// W8A8SparseCompressedLinear_57088705298809
// MI455X (gfx1250) — 8-bit matrix operand order repaired and checked by reading, compiled, not run on hardware
//
#include <hip/hip_runtime.h>
#include <hip/hip_bf16.h>
#include <cstdint>

#define MDIM 8192
#define KDIM 4096
#define NDIM 4096

#define KSTAGE 128
#define LDS_STRIDE 144
#define TILE_BYTES (128 * LDS_STRIDE)

typedef __attribute__((ext_vector_type(8))) int v8i;
typedef int v4i_raw __attribute__((vector_size(16)));

#define HAS_ASYNC_LDS __has_builtin(__builtin_amdgcn_global_load_async_to_lds_b128)
#define HAS_WAIT_ASYNC __has_builtin(__builtin_amdgcn_s_wait_asynccnt)

__device__ __forceinline__ int quant1(float f, float inv, float off) {
  float q = rintf(f * inv) + off;
  q = fminf(fmaxf(q, -128.0f), 127.0f);
  return (int)q;
}

__device__ __forceinline__ unsigned pack4(float4 v, float inv, float off) {
  unsigned a = (unsigned)(quant1(v.x, inv, off) & 0xFF);
  unsigned b = (unsigned)(quant1(v.y, inv, off) & 0xFF);
  unsigned c = (unsigned)(quant1(v.z, inv, off) & 0xFF);
  unsigned d = (unsigned)(quant1(v.w, inv, off) & 0xFF);
  return a | (b << 8) | (c << 16) | (d << 24);
}

__global__ __launch_bounds__(256)
void quantize_x_kernel(const float* __restrict__ x,
                       const float* __restrict__ s,
                       const int* __restrict__ o,
                       uint32_t* __restrict__ xq) {
  int i = blockIdx.x * blockDim.x + threadIdx.x;
  float inv = 1.0f / s[0];
  float off = (float)o[0];
  const float4* src = (const float4*)x + (size_t)i * 4;
  float4 v0 = src[0], v1 = src[1], v2 = src[2], v3 = src[3];
  uint4 r;
  r.x = pack4(v0, inv, off);
  r.y = pack4(v1, inv, off);
  r.z = pack4(v2, inv, off);
  r.w = pack4(v3, inv, off);
  ((uint4*)xq)[i] = r;
}

__global__ __launch_bounds__(256)
void repack_w_kernel(const int* __restrict__ w,
                     const int* __restrict__ index,
                     const float* __restrict__ dscale,
                     const int* __restrict__ qbias,
                     int8_t* __restrict__ wp,
                     float* __restrict__ sp,
                     int* __restrict__ bp) {
  const int j = blockIdx.x;
  const int t = threadIdx.x;
  const int src = index[j];
  if (t == 0) {
    sp[j] = dscale[src];
    bp[j] = qbias[src];
  }
  const int4* wr = (const int4*)(w + (size_t)src * KDIM);
  uint32_t* outp = (uint32_t*)(wp + (size_t)j * KDIM);
#pragma unroll
  for (int r = 0; r < 4; ++r) {
    int idx = t + r * 256;
    int4 v = wr[idx];
    outp[idx] = (unsigned)(v.x & 0xFF) | ((unsigned)(v.y & 0xFF) << 8) |
                ((unsigned)(v.z & 0xFF) << 16) | ((unsigned)(v.w & 0xFF) << 24);
  }
}

__device__ __forceinline__ void cp16_g2l(int8_t* lds_dst, const int8_t* gsrc) {
#if HAS_ASYNC_LDS
  __builtin_amdgcn_global_load_async_to_lds_b128(
      (__attribute__((address_space(1))) v4i_raw*)gsrc,
      (__attribute__((address_space(3))) v4i_raw*)lds_dst,
      0, 0);
#else
  *(int4*)lds_dst = *(const int4*)gsrc;
#endif
}

__device__ __forceinline__ void wait_async_8() {
#if HAS_WAIT_ASYNC
  __builtin_amdgcn_s_wait_asynccnt(8);
#elif HAS_ASYNC_LDS
  asm volatile("s_wait_asynccnt 0x8" ::: "memory");
#endif
}

__device__ __forceinline__ void wait_async_0() {
#if HAS_WAIT_ASYNC
  __builtin_amdgcn_s_wait_asynccnt(0);
#elif HAS_ASYNC_LDS
  asm volatile("s_wait_asynccnt 0x0" ::: "memory");
#endif
}

__device__ __forceinline__ void stage_in(const int8_t* __restrict__ Aq,
                                         const int8_t* __restrict__ Bp,
                                         int8_t* ldsA, int8_t* ldsB,
                                         int mBlock, int nBlock, int k0, int tid) {
#pragma unroll
  for (int r = 0; r < 4; ++r) {
    int o   = tid + r * 256;
    int row = o >> 3;
    int c   = (o & 7) * 16;
    cp16_g2l(ldsA + row * LDS_STRIDE + c,
             Aq + (size_t)(mBlock + row) * KDIM + k0 + c);
    cp16_g2l(ldsB + row * LDS_STRIDE + c,
             Bp + (size_t)(nBlock + row) * KDIM + k0 + c);
  }
}

__global__ __launch_bounds__(256)
void gemm_w8a8_kernel(const int8_t* __restrict__ Aq,
                      const int8_t* __restrict__ Bp,
                      const float* __restrict__ sp,
                      const int* __restrict__ bp,
                      float* __restrict__ out) {
  __shared__ int8_t smem[2][2][TILE_BYTES];

  const int tid  = threadIdx.x;
  const int lane = tid & 31;
  const int wave = tid >> 5;
  const int wm = wave >> 2;
  const int wn = wave & 3;
  const int l16 = lane & 15;
  const int lh  = lane >> 4;

  const int mBlock = blockIdx.y * 128;
  const int nBlock = blockIdx.x * 128;

  v8i acc[4][2];
#pragma unroll
  for (int i = 0; i < 4; ++i)
#pragma unroll
    for (int j = 0; j < 2; ++j)
      acc[i][j] = (v8i)0;

  const int aBase[4] = {
      (wm * 64 +  0 + l16) * LDS_STRIDE + 8 * lh,
      (wm * 64 + 16 + l16) * LDS_STRIDE + 8 * lh,
      (wm * 64 + 32 + l16) * LDS_STRIDE + 8 * lh,
      (wm * 64 + 48 + l16) * LDS_STRIDE + 8 * lh };
  const int bBase[2] = {
      (wn * 32 +  0 + l16) * LDS_STRIDE + 8 * lh,
      (wn * 32 + 16 + l16) * LDS_STRIDE + 8 * lh };

  constexpr int NSTAGES = KDIM / KSTAGE;

  int8_t* aCur = &smem[0][0][0];
  int8_t* bCur = &smem[0][1][0];
  int8_t* aNxt = &smem[1][0][0];
  int8_t* bNxt = &smem[1][1][0];

  stage_in(Aq, Bp, aCur, bCur, mBlock, nBlock, 0, tid);

#pragma unroll 1
  for (int s = 0; s < NSTAGES; ++s) {
    if (s + 1 < NSTAGES) {
      stage_in(Aq, Bp, aNxt, bNxt, mBlock, nBlock, (s + 1) * KSTAGE, tid);
      wait_async_8();
    } else {
      wait_async_0();
    }
    __syncthreads();

#pragma unroll
    for (int kk = 0; kk < KSTAGE; kk += 64) {
      v8i b[2];
#pragma unroll
      for (int j = 0; j < 2; ++j) {
        const int8_t* p = bCur + bBase[j] + kk;
        int2 u0 = *(const int2*)(p);
        int2 u1 = *(const int2*)(p + 16);
        int2 u2 = *(const int2*)(p + 32);
        int2 u3 = *(const int2*)(p + 48);
        b[j][0] = u0.x; b[j][1] = u0.y; b[j][2] = u1.x; b[j][3] = u1.y;
        b[j][4] = u2.x; b[j][5] = u2.y; b[j][6] = u3.x; b[j][7] = u3.y;
      }
      v8i a[4];
#pragma unroll
      for (int i = 0; i < 4; ++i) {
        const int8_t* p = aCur + aBase[i] + kk;
        int2 t0 = *(const int2*)(p);
        int2 t1 = *(const int2*)(p + 16);
        int2 t2 = *(const int2*)(p + 32);
        int2 t3 = *(const int2*)(p + 48);
        a[i][0] = t0.x; a[i][1] = t0.y; a[i][2] = t1.x; a[i][3] = t1.y;
        a[i][4] = t2.x; a[i][5] = t2.y; a[i][6] = t3.x; a[i][7] = t3.y;
      }
#pragma unroll
      for (int i = 0; i < 4; ++i)
#pragma unroll
        for (int j = 0; j < 2; ++j)
          acc[i][j] = __builtin_amdgcn_wmma_i32_16x16x64_iu8(
               true, a[i],  true, b[j],
              acc[i][j],  false,  false);
    }

    __syncthreads();

    int8_t* ta = aCur; aCur = aNxt; aNxt = ta;
    int8_t* tb = bCur; bCur = bNxt; bNxt = tb;
  }

#pragma unroll
  for (int j = 0; j < 2; ++j) {
    int n = nBlock + wn * 32 + j * 16 + l16;
    float s = sp[n];
    int   bb = bp[n];
#pragma unroll
    for (int i = 0; i < 4; ++i) {
      int mBase = mBlock + wm * 64 + i * 16 + lh * 8;
      float* po = out + (size_t)mBase * NDIM + n;
#pragma unroll
      for (int r = 0; r < 8; ++r)
        po[(size_t)r * NDIM] = (float)(acc[i][j][r] + bb) * s;
    }
  }
}

extern "C" void kernel_launch(void* const* d_in, const int* in_sizes, int n_in,
                              void* d_out, int out_size, void* d_ws, size_t ws_size,
                              hipStream_t stream) {
  const float* x            = (const float*)d_in[0];
  const int*   weight       = (const int*)d_in[1];
  const float* input_scale  = (const float*)d_in[2];
  const int*   input_offset = (const int*)d_in[3];
  const float* deq_scale    = (const float*)d_in[4];
  const int*   quant_bias   = (const int*)d_in[5];
  const int*   index        = (const int*)d_in[6];
  float* out = (float*)d_out;

  int8_t* xq  = (int8_t*)d_ws;
  int8_t* wp  = xq + (size_t)MDIM * KDIM;
  float*  spv = (float*)(wp + (size_t)NDIM * KDIM);
  int*    bpv = (int*)(spv + NDIM);

  {
    int total = MDIM * KDIM / 16;
    quantize_x_kernel<<<total / 256, 256, 0, stream>>>(x, input_scale, input_offset,
                                                       (uint32_t*)xq);
  }
  repack_w_kernel<<<NDIM, 256, 0, stream>>>(weight, index, deq_scale, quant_bias,
                                            wp, spv, bpv);
  {
    dim3 grid(NDIM / 128, MDIM / 128);
    gemm_w8a8_kernel<<<grid, 256, 0, stream>>>(xq, wp, spv, bpv, out);
  }
}
